// MemoryController_35648228557105
// MI455X (gfx1250) — compile-verified, matrix operand order checked by reading, not run on hardware
//
#include <hip/hip_runtime.h>
#include <hip/hip_bf16.h>

// Problem constants (from reference)
#define B_  32
#define T_  256
#define D_  1024
#define M_  256
#define S_  64

// padded LDS row strides (avoid 16-way bank conflicts on A-fragment loads)
#define WS32 260   // f32 row stride for s_mem   (260 % 64 == 4)
#define WSH  264   // f16 row stride for s_memh / s_rmh (264/2=132 dwords, %64 == 4)

typedef __attribute__((ext_vector_type(16))) _Float16 v16h;
typedef __attribute__((ext_vector_type(8)))  float    v8f;
typedef __attribute__((ext_vector_type(4)))  unsigned u32x4;

union ABu { v16h v; u32x4 q[2]; };

// Two 16B loads per fragment: source halfs [0..8) and [16..24) past p.
__device__ __forceinline__ v16h load_frag(const _Float16* p) {
  ABu t;
  const u32x4* q = (const u32x4*)p;
  t.q[0] = q[0];
  t.q[1] = q[2];
  return t.v;
}

#define WMMA(a, b, c) __builtin_amdgcn_wmma_f32_16x16x32_f16(false, (a), false, (b), (short)0, (c), false, false)

// ---------------- workspace layout (all _Float16) -------------------------
// Wiv : [512][1024]  rows 0..255 = Wi, 256..511 = Wv
// Wxs : [768][256]   rows = Wr[:, :256] ++ Wz[:, :256] ++ Wu[:, :256]
// Whr/Whz/Whu : [256][256]  W*[:, 256:512]
#define WIV_OFF  0
#define WXS_OFF  (512*1024)
#define WHR_OFF  (WXS_OFF + 768*256)
#define WHZ_OFF  (WHR_OFF + 256*256)
#define WHU_OFF  (WHZ_OFF + 256*256)

__global__ void prep_weights(const float* __restrict__ Wi, const float* __restrict__ Wv,
                             const float* __restrict__ Wr, const float* __restrict__ Wz,
                             const float* __restrict__ Wu, _Float16* __restrict__ ws) {
  const int stride = gridDim.x * blockDim.x;
  const int tid0 = blockIdx.x * blockDim.x + threadIdx.x;
  _Float16* Wiv = ws + WIV_OFF;
  _Float16* Wxs = ws + WXS_OFF;
  _Float16* Whr = ws + WHR_OFF;
  _Float16* Whz = ws + WHZ_OFF;
  _Float16* Whu = ws + WHU_OFF;
  for (int i = tid0; i < 512*1024; i += stride) {
    int r = i >> 10, c = i & 1023;
    float v = (r < 256) ? Wi[r*1024 + c] : Wv[(r-256)*1024 + c];
    Wiv[i] = (_Float16)v;
  }
  for (int i = tid0; i < 768*256; i += stride) {
    int r = i >> 8, c = i & 255;
    const float* W = (r < 256) ? Wr : ((r < 512) ? Wz : Wu);
    Wxs[i] = (_Float16)W[(r & 255)*512 + c];
  }
  for (int i = tid0; i < 256*256; i += stride) {
    int r = i >> 8, c = i & 255;
    Whr[i] = (_Float16)Wr[r*512 + 256 + c];
    Whz[i] = (_Float16)Wz[r*512 + 256 + c];
    Whu[i] = (_Float16)Wu[r*512 + 256 + c];
  }
}

// Dynamic LDS carve (bytes):
//   f32:  s_mem[64*WS32], s_iv[512], s_vxs[768], s_biv[512], s_bxs[768],
//         s_ww[64], s_usage[64], s_sc[64]
//   f16:  s_memh[64*WSH], s_rmh[64*WSH], s_xh[1024], s_valsh[256], s_zero[32]
#define SMEM_BYTES ((64*WS32 + 512 + 768 + 512 + 768 + 64 + 64 + 64) * 4 + \
                    (2*64*WSH + 1024 + 256 + 32) * 2)

__global__ __launch_bounds__(256) void mc_persistent(
    const float* __restrict__ hs,       // [B,T,D]
    const float* __restrict__ meminit,  // [B,S,M]
    const float* __restrict__ bi, const float* __restrict__ bv,
    const float* __restrict__ br, const float* __restrict__ bz,
    const float* __restrict__ bu,
    const _Float16* __restrict__ ws,
    float* __restrict__ out)            // [B,S,M]
{
  extern __shared__ char smem_raw[];
  float*    s_mem   = (float*)smem_raw;             // [64][WS32]
  float*    s_iv    = s_mem + 64*WS32;              // [512]  mem_input ++ vals
  float*    s_vxs   = s_iv + 512;                   // [768]  vr ++ vz ++ vu
  float*    s_biv   = s_vxs + 768;                  // [512]  bi ++ bv
  float*    s_bxs   = s_biv + 512;                  // [768]  br ++ bz ++ bu
  float*    s_ww    = s_bxs + 768;                  // [64]
  float*    s_usage = s_ww + 64;                    // [64]
  float*    s_sc    = s_usage + 64;                 // [64]
  _Float16* s_memh  = (_Float16*)(s_sc + 64);       // [64][WSH]
  _Float16* s_rmh   = s_memh + 64*WSH;              // [64][WSH]  r * mem
  _Float16* s_xh    = s_rmh + 64*WSH;               // [1024]
  _Float16* s_valsh = s_xh + 1024;                  // [256]
  _Float16* s_zero  = s_valsh + 256;                // [32] stays zero

  const int b    = blockIdx.x;
  const int tid  = threadIdx.x;
  const int lane = tid & 31;
  const int wv   = tid >> 5;       // wave id 0..7
  const int nl   = lane & 15;      // N/M index within tile
  const int hb   = lane >> 4;      // K-half select
  const int kb   = hb * 8;         // K base offset (elements)

  const _Float16* Wiv = ws + WIV_OFF;
  const _Float16* Wxs = ws + WXS_OFF;
  const _Float16* Whr = ws + WHR_OFF;
  const _Float16* Whz = ws + WHZ_OFF;
  const _Float16* Whu = ws + WHU_OFF;

  // ---- init: pad, biases, x_0, mem = normalize(init*0.01), usage=0 ----
  if (tid < 32) s_zero[tid] = (_Float16)0.f;
  for (int i = tid; i < 512; i += 256) s_biv[i] = (i < 256) ? bi[i] : bv[i-256];
  for (int i = tid; i < 768; i += 256)
    s_bxs[i] = (i < 256) ? br[i] : ((i < 512) ? bz[i-256] : bu[i-512]);
  {
    const float* x0 = hs + (size_t)b * T_ * D_;
    #pragma unroll
    for (int j = 0; j < 4; ++j) s_xh[tid + j*256] = (_Float16)x0[tid + j*256];
  }
  for (int s = wv; s < S_; s += 8) {
    float vbuf[8];
    float ss = 0.f;
    #pragma unroll
    for (int j = 0; j < 8; ++j) {
      float v = meminit[(b*S_ + s)*M_ + lane + j*32] * 0.01f;
      vbuf[j] = v; ss += v*v;
    }
    for (int off = 16; off; off >>= 1) ss += __shfl_xor(ss, off, 32);
    float scale = 1.f / fmaxf(sqrtf(ss), 1e-12f);
    #pragma unroll
    for (int j = 0; j < 8; ++j) {
      float v = vbuf[j] * scale;
      s_mem[s*WS32 + lane + j*32] = v;
      s_memh[s*WSH + lane + j*32] = (_Float16)v;
    }
  }
  if (tid < S_) s_usage[tid] = 0.f;
  float age = 0.f;
  __syncthreads();

  for (int t = 0; t < T_; ++t) {
    // ---- 0. issue loads for x_{t+1}; latency hidden under this step ----
    float xr[4];
    {
      const int tn = (t + 1 < T_) ? (t + 1) : t;
      const float* xnext = hs + (size_t)(b*T_ + tn) * D_;
      #pragma unroll
      for (int j = 0; j < 4; ++j) xr[j] = xnext[tid + j*256];
    }

    // ---- 2. GEMV1 via WMMA: iv[0:512] = x @ [Wi;Wv]^T ----
    // Ping-pong (modulo-2) software pipeline: no rotating-register copies.
    {
      v8f acc[4];
      #pragma unroll
      for (int j = 0; j < 4; ++j) acc[j] = (v8f){};
      const _Float16* ab = s_xh + kb;                          // valid rows only on nl==0
      const _Float16* bbase = Wiv + (wv*16 + nl)*1024 + kb;    // tile jj at +jj*8*16*1024
      v16h a0, a1, b0[4], b1[4];
      a0 = load_frag((nl == 0) ? ab : s_zero);
      #pragma unroll
      for (int jj = 0; jj < 4; ++jj) b0[jj] = load_frag(bbase + jj*(8*16*1024));
      for (int k0 = 0; k0 < 1024; k0 += 64) {
        // prefetch k0+32 into buffer 1
        a1 = load_frag((nl == 0) ? (ab + k0 + 32) : s_zero);
        #pragma unroll
        for (int jj = 0; jj < 4; ++jj)
          b1[jj] = load_frag(bbase + jj*(8*16*1024) + k0 + 32);
        #pragma unroll
        for (int jj = 0; jj < 4; ++jj) acc[jj] = WMMA(a0, b0[jj], acc[jj]);
        // prefetch k0+64 into buffer 0 (dead reload on final iteration)
        const int kn = (k0 + 64 < 1024) ? (k0 + 64) : k0;
        a0 = load_frag((nl == 0) ? (ab + kn) : s_zero);
        #pragma unroll
        for (int jj = 0; jj < 4; ++jj)
          b0[jj] = load_frag(bbase + jj*(8*16*1024) + kn);
        #pragma unroll
        for (int jj = 0; jj < 4; ++jj) acc[jj] = WMMA(a1, b1[jj], acc[jj]);
      }
      if (lane < 16) {
        #pragma unroll
        for (int jj = 0; jj < 4; ++jj) {
          int n2 = (wv + 8*jj)*16 + lane;
          s_iv[n2] = acc[jj][0] + s_biv[n2];
        }
      }
    }
    __syncthreads();

    // ---- 3. sim -> softmax; vals -> f16; stash x_{t+1} into s_xh ----
    #pragma unroll
    for (int j = 0; j < 4; ++j) s_xh[tid + j*256] = (_Float16)xr[j];
    if (tid < 64) {
      float acc = 0.f;
      for (int m = 0; m < M_; ++m) acc += s_iv[m] * s_mem[tid*WS32 + m];
      float score = acc - age*0.1f - s_usage[tid]*0.2f;
      s_sc[tid] = -score;                           // softmax(-scores)
    } else if (tid < 192) {
      int i = tid - 64;
      s_valsh[2*i]   = (_Float16)s_iv[256 + 2*i];
      s_valsh[2*i+1] = (_Float16)s_iv[256 + 2*i + 1];
    }
    __syncthreads();
    if (tid < 64) {
      float mx = -1e30f;
      for (int s = 0; s < 64; ++s) mx = fmaxf(mx, s_sc[s]);
      float sum = 0.f;
      for (int s = 0; s < 64; ++s) sum += __expf(s_sc[s] - mx);
      s_ww[tid] = __expf(s_sc[tid] - mx) / sum;
    }
    __syncthreads();

    // ---- 4. GEMV2 via WMMA: vxs[0:768] = vals @ [Wr_x;Wz_x;Wu_x]^T + bias ----
    {
      v8f acc[6];
      #pragma unroll
      for (int j = 0; j < 6; ++j) acc[j] = (v8f){};
      const _Float16* ab = s_valsh + kb;
      const _Float16* bbase = Wxs + (wv*16 + nl)*256 + kb;     // tile jj at +jj*8*16*256
      #pragma unroll
      for (int k0 = 0; k0 < 256; k0 += 32) {
        v16h a = load_frag((nl == 0) ? (ab + k0) : s_zero);
        #pragma unroll
        for (int jj = 0; jj < 6; ++jj)
          acc[jj] = WMMA(a, load_frag(bbase + jj*(8*16*256) + k0), acc[jj]);
      }
      if (lane < 16) {
        #pragma unroll
        for (int jj = 0; jj < 6; ++jj) {
          int n2 = (wv + 8*jj)*16 + lane;
          s_vxs[n2] = acc[jj][0] + s_bxs[n2];
        }
      }
    }
    __syncthreads();

    // Fixed per-wave tiling for the slot GEMMs: wave wv covers rows
    // [mt*16, mt*16+16) and the 8 N-tiles starting at ntb.
    const int mt  = wv >> 1;
    const int ntb = (wv & 1) * 8;

    // ---- 5. r GEMM: r = sigmoid(mem @ Whr^T + vr); stash f16(r*mem) ----
    {
      v8f acc[8];
      #pragma unroll
      for (int i = 0; i < 8; ++i) acc[i] = (v8f){};
      const _Float16* ap = s_memh + (mt*16 + nl)*WSH + kb;
      const _Float16* bbase = Whr + (ntb*16 + nl)*256 + kb;    // tile i at +i*16*256
      #pragma unroll
      for (int k0 = 0; k0 < 256; k0 += 32) {
        v16h a = load_frag(ap + k0);
        #pragma unroll
        for (int i = 0; i < 8; ++i)
          acc[i] = WMMA(a, load_frag(bbase + i*(16*256) + k0), acc[i]);
      }
      #pragma unroll
      for (int i = 0; i < 8; ++i) {
        int N = (ntb + i)*16 + nl;
        float vb = s_vxs[N];
        #pragma unroll
        for (int r_ = 0; r_ < 8; ++r_) {
          int Mrow = mt*16 + r_ + hb*8;
          float r = 1.f / (1.f + __expf(-(acc[i][r_] + vb)));
          s_rmh[Mrow*WSH + N] = (_Float16)(r * s_mem[Mrow*WS32 + N]);
        }
      }
    }
    __syncthreads();

    // ---- 6. fused z + cand GEMMs, blend + masked write (pre-norm) ----
    {
      v8f cz[8], cu[8];
      #pragma unroll
      for (int i = 0; i < 8; ++i) { cz[i] = (v8f){}; cu[i] = (v8f){}; }
      const _Float16* apm = s_memh + (mt*16 + nl)*WSH + kb;
      const _Float16* apr = s_rmh  + (mt*16 + nl)*WSH + kb;
      const _Float16* bzb = Whz + (ntb*16 + nl)*256 + kb;
      const _Float16* bub = Whu + (ntb*16 + nl)*256 + kb;
      #pragma unroll
      for (int k0 = 0; k0 < 256; k0 += 32) {
        v16h a1 = load_frag(apm + k0);
        v16h a2 = load_frag(apr + k0);
        #pragma unroll
        for (int i = 0; i < 8; ++i) {
          cz[i] = WMMA(a1, load_frag(bzb + i*(16*256) + k0), cz[i]);
          cu[i] = WMMA(a2, load_frag(bub + i*(16*256) + k0), cu[i]);
        }
      }
      #pragma unroll
      for (int i = 0; i < 8; ++i) {
        int N = (ntb + i)*16 + nl;
        float vbz = s_vxs[256 + N];
        float vbu = s_vxs[512 + N];
        #pragma unroll
        for (int r_ = 0; r_ < 8; ++r_) {
          int Mrow = mt*16 + r_ + hb*8;
          float h    = s_mem[Mrow*WS32 + N];
          float z    = 1.f / (1.f + __expf(-(cz[i][r_] + vbz)));
          float cand = tanhf(cu[i][r_] + vbu);
          float nh   = (1.f - z)*h + z*cand;
          float wwv  = s_ww[Mrow];
          float w    = wwv * 0.5f;                  // UPDATE_RATE
          s_mem[Mrow*WS32 + N] = (wwv > 0.01f) ? (h*(1.f - w) + nh*w) : h;
        }
      }
    }
    __syncthreads();

    // ---- 7. per-slot renorm + usage/age ----
    for (int s = wv; s < S_; s += 8) {
      float ss = 0.f;
      #pragma unroll
      for (int j = 0; j < 8; ++j) { float v = s_mem[s*WS32 + lane + j*32]; ss += v*v; }
      for (int off = 16; off; off >>= 1) ss += __shfl_xor(ss, off, 32);
      float scale = 1.f / fmaxf(sqrtf(ss), 1e-12f);
      #pragma unroll
      for (int j = 0; j < 8; ++j) {
        float v = s_mem[s*WS32 + lane + j*32] * scale;
        s_mem[s*WS32 + lane + j*32] = v;
        s_memh[s*WSH + lane + j*32] = (_Float16)v;
      }
    }
    if (tid < 64) {
      float wwv = s_ww[tid];
      s_usage[tid] = (s_usage[tid] + (wwv > 0.01f ? wwv : 0.f)) * 0.99f;
    }
    age = age * 0.98f + 1.f;
    __syncthreads();
  }

  // ---- write final memory (un-pad rows) ----
  for (int i = tid; i < S_*M_; i += 256) {
    int s = i >> 8, m = i & 255;
    out[(size_t)b*S_*M_ + i] = s_mem[s*WS32 + m];
  }
}

extern "C" void kernel_launch(void* const* d_in, const int* in_sizes, int n_in,
                              void* d_out, int out_size, void* d_ws, size_t ws_size,
                              hipStream_t stream) {
  const float* hs      = (const float*)d_in[0];
  const float* meminit = (const float*)d_in[1];
  const float* Wi = (const float*)d_in[2];
  const float* bi = (const float*)d_in[3];
  const float* Wv = (const float*)d_in[4];
  const float* bv = (const float*)d_in[5];
  const float* Wr = (const float*)d_in[6];
  const float* br = (const float*)d_in[7];
  const float* Wz = (const float*)d_in[8];
  const float* bz = (const float*)d_in[9];
  const float* Wu = (const float*)d_in[10];
  const float* bu = (const float*)d_in[11];
  _Float16* ws = (_Float16*)d_ws;

  prep_weights<<<256, 256, 0, stream>>>(Wi, Wv, Wr, Wz, Wu, ws);
  mc_persistent<<<B_, 256, SMEM_BYTES, stream>>>(hs, meminit, bi, bv, br, bz, bu,
                                                 ws, (float*)d_out);
}
